// INTagger_64347200028806
// MI455X (gfx1250) — matrix operand order repaired and checked by reading, not compiled, not run on hardware
//
#include <hip/hip_runtime.h>
#include <hip/hip_bf16.h>
#include <stdint.h>

// Problem dims (from reference)
#define B_    128
#define N_    64
#define NV_   14
#define P_    32
#define S_    14
#define H_    128
#define DE_   64
#define DO_   64
#define NCLS_ 5

typedef __attribute__((ext_vector_type(16))) __bf16          v16bf;
typedef __attribute__((ext_vector_type(8)))  float           v8f;
typedef __attribute__((ext_vector_type(8)))  unsigned short  v8us;
typedef __attribute__((ext_vector_type(16))) unsigned short  v16us;

__device__ __forceinline__ unsigned short f2bf(float f) {
  return __builtin_bit_cast(unsigned short, (__bf16)f);   // RTNE -> v_cvt_pk_bf16_f32
}

// ReLU in one instruction: v_med3_num_f32(x, 0, +inf)
__device__ __forceinline__ float relu(float x) {
  return __builtin_amdgcn_fmed3f(x, 0.f, __builtin_inff());
}

__device__ __forceinline__ v16bf mkfrag(v8us lo, v8us hi) {
  v16us a;
#pragma unroll
  for (int i = 0; i < 8; i++) { a[i] = lo[i]; a[i + 8] = hi[i]; }
  return __builtin_bit_cast(v16bf, a);
}

// ---- WMMA operand assembly (wave32, v_wmma_f32_16x16x32_bf16) ----
// A operand = WEIGHTS, stored transposed in LDS: Wt[out][in] (ld = ldk).
//   A-layout: lane m=lane&15 -> output row nt*16+m; elems 0-7 -> K=kb..kb+7,
//   8-15 -> K=kb+16..kb+23, kb=(lane>>4)*8  => two 16B row-contiguous loads.
__device__ __forceinline__ v16bf afrag_w(const unsigned short* Wt, int ldk,
                                         int nt, int ks, int lane) {
  const unsigned short* p = Wt + (nt * 16 + (lane & 15)) * ldk + ks * 32 + ((lane >> 4) << 3);
  return mkfrag(*(const v8us*)p, *(const v8us*)(p + 16));
}

// B operand = ACTIVATION row (row-major, bf16): lane n=lane&15 -> row n;
//   elems 0-7 -> K = ks*32 + (lane>>4)*8 + j, 8-15 -> +16  => two 16B loads.
__device__ __forceinline__ v16bf bfrag_r(const unsigned short* rowp, int ks, int lane) {
  const unsigned short* p = rowp + ks * 32 + ((lane >> 4) << 3);
  return mkfrag(*(const v8us*)p, *(const v8us*)(p + 16));
}

__device__ __forceinline__ v8f wmma_bf16(v16bf a, v16bf b, v8f c) {
  return __builtin_amdgcn_wmma_f32_16x16x32_bf16(false, a, false, b, (short)0, c, false, false);
}

// D layout: VGPR i -> feature nt*16 + (lane>>4)*8 + i, lane&15 -> row.
// => lane's 8 outputs are feature-contiguous: ReLU + pack -> one b128 store.
__device__ __forceinline__ void store_row8_relu(unsigned short* dst, v8f acc) {
  v8us o;
#pragma unroll
  for (int i = 0; i < 8; i++) o[i] = f2bf(relu(acc[i]));
  *(v8us*)dst = o;
}

// ---- layers 2+3 (shared by all three MLPs): 128 -> 128 -> 64, ReLU ----
__device__ __forceinline__ void mlp_l23(
    unsigned short* act1, unsigned short* act2,          // per-wave [16][136]
    const unsigned short* W2t, const float* b2,
    const unsigned short* W3t, const float* b3,
    int lane, v8f* out)
{
  const int LDA = 136;
  const int n  = lane & 15;
  const int mb = (lane >> 4) << 3;
  const unsigned short* row1 = act1 + n * LDA;
#pragma unroll
  for (int nt = 0; nt < 8; nt++) {
    v8f acc = *(const v8f*)(b2 + nt * 16 + mb);
#pragma unroll
    for (int ks = 0; ks < 4; ks++)
      acc = wmma_bf16(afrag_w(W2t, 128, nt, ks, lane), bfrag_r(row1, ks, lane), acc);
    store_row8_relu(act2 + n * LDA + nt * 16 + mb, acc);
  }
  const unsigned short* row2 = act2 + n * LDA;
#pragma unroll
  for (int nt = 0; nt < 4; nt++) {
    v8f acc = *(const v8f*)(b3 + nt * 16 + mb);
#pragma unroll
    for (int ks = 0; ks < 4; ks++)
      acc = wmma_bf16(afrag_w(W3t, 128, nt, ks, lane), bfrag_r(row2, ks, lane), acc);
#pragma unroll
    for (int i = 0; i < 8; i++) acc[i] = relu(acc[i]);
    out[nt] = acc;
  }
}

// ---- cooperative loaders ----
__device__ __forceinline__ void load_wt(unsigned short* Wt, const float* w,
                                        int fin, int finPad, int fout, int tid) {
  int total = fout * finPad;
  for (int i = tid; i < total; i += 256) {
    int n = i / finPad;
    int k = i - n * finPad;
    Wt[i] = (k < fin) ? f2bf(w[k * fout + n]) : (unsigned short)0;
  }
}
__device__ __forceinline__ void load_vec(float* dst, const float* src, int n, int tid) {
  for (int i = tid; i < n; i += 256) dst[i] = src[i];
}

// =================== zero workspace ===================
__global__ void intagger_zero_kernel(float* p, int n) {
  int i = blockIdx.x * blockDim.x + threadIdx.x;
  if (i < n) p[i] = 0.f;
}

// =================== kernel A: particle-particle edges ===================
// grid (B, 4); 4032 edges/batch = 252 tiles of 16; 63 tiles per chunk; 8 waves
__global__ __launch_bounds__(256) void intagger_pp_kernel(
    const float* __restrict__ x,
    const float* __restrict__ w1, const float* __restrict__ g1,
    const float* __restrict__ w2, const float* __restrict__ g2,
    const float* __restrict__ w3, const float* __restrict__ g3,
    float* __restrict__ epp)
{
  extern __shared__ unsigned short smem[];
  unsigned short* W1t  = smem;                      // 128*64
  unsigned short* W2t  = W1t + 128 * 64;            // 128*128
  unsigned short* W3t  = W2t + 128 * 128;           // 64*128
  unsigned short* x_sh = W3t + 64 * 128;            // [64][32] bf16 xT
  unsigned short* stg  = x_sh + 64 * 32;            // 8 waves * 2*(16*136)
  float* acc_sh = (float*)(stg + 8 * 2 * 16 * 136); // [64][64] f32 accumulator
  float* b1 = acc_sh + 64 * 64;
  float* b2 = b1 + 128;
  float* b3 = b2 + 128;

  int tid = threadIdx.x, lane = tid & 31, wave = tid >> 5;
  int bb = blockIdx.x, chunk = blockIdx.y;

  load_wt(W1t, w1, 64, 64, 128, tid);
  load_wt(W2t, w2, 128, 128, 128, tid);
  load_wt(W3t, w3, 128, 128, 64, tid);
  load_vec(b1, g1, 128, tid);
  load_vec(b2, g2, 128, tid);
  load_vec(b3, g3, 64, tid);
  for (int i = tid; i < 64 * 32; i += 256) {        // xT[n][p]
    int n = i >> 5, p = i & 31;
    x_sh[i] = f2bf(x[(size_t)bb * (P_ * N_) + p * N_ + n]);
  }
  for (int i = tid; i < 64 * 64; i += 256) acc_sh[i] = 0.f;
  __syncthreads();

  unsigned short* act1 = stg + wave * (2 * 16 * 136);
  unsigned short* act2 = act1 + 16 * 136;
  int n  = lane & 15;
  int mb = (lane >> 4) << 3;

  for (int t = chunk * 63 + wave; t < chunk * 63 + 63; t += 8) {
    // per-lane row: e -> (receiver r, sender s); layer-1 B rows straight from x_sh
    int e = t * 16 + n;
    int r = e / 63, s0 = e - r * 63;
    int s = s0 + (s0 >= r);
    const unsigned short* rowR = x_sh + r * 32;     // K 0..31
    const unsigned short* rowS = x_sh + s * 32;     // K 32..63
#pragma unroll
    for (int nt = 0; nt < 8; nt++) {
      v8f acc = *(const v8f*)(b1 + nt * 16 + mb);
      acc = wmma_bf16(afrag_w(W1t, 64, nt, 0, lane), bfrag_r(rowR, 0, lane), acc);
      acc = wmma_bf16(afrag_w(W1t, 64, nt, 1, lane), bfrag_r(rowS, 0, lane), acc);
      store_row8_relu(act1 + n * 136 + nt * 16 + mb, acc);
    }
    v8f out[4];
    mlp_l23(act1, act2, W2t, b2, W3t, b3, lane, out);

    // scatter-add into LDS accumulator (ds_add_f32), receiver = r
#pragma unroll
    for (int nt = 0; nt < 4; nt++) {
      float* base = acc_sh + r * 64 + nt * 16 + mb;
#pragma unroll
      for (int i = 0; i < 8; i++) atomicAdd(base + i, out[nt][i]);
    }
  }
  __syncthreads();
  for (int i = tid; i < 64 * 64; i += 256)
    unsafeAtomicAdd(&epp[(size_t)bb * (N_ * DE_) + i], acc_sh[i]);
}

// =================== kernel B: particle-vertex edges ===================
// grid (B, 2); 896 edges/batch = 56 tiles; 28 per chunk; K1 = 46 padded to 64
__global__ __launch_bounds__(256) void intagger_pv_kernel(
    const float* __restrict__ x, const float* __restrict__ y,
    const float* __restrict__ w1, const float* __restrict__ g1,
    const float* __restrict__ w2, const float* __restrict__ g2,
    const float* __restrict__ w3, const float* __restrict__ g3,
    float* __restrict__ epv)
{
  extern __shared__ unsigned short smem[];
  unsigned short* W1t   = smem;
  unsigned short* W2t   = W1t + 128 * 64;
  unsigned short* W3t   = W2t + 128 * 128;
  unsigned short* x_sh  = W3t + 64 * 128;           // [64][32]
  unsigned short* y_pad = x_sh + 64 * 32;           // [14][32], cols 14..31 zero (alloc 512)
  unsigned short* stg   = y_pad + 512;
  float* acc_sh = (float*)(stg + 8 * 2 * 16 * 136);
  float* b1 = acc_sh + 64 * 64;
  float* b2 = b1 + 128;
  float* b3 = b2 + 128;

  int tid = threadIdx.x, lane = tid & 31, wave = tid >> 5;
  int bb = blockIdx.x, chunk = blockIdx.y;

  load_wt(W1t, w1, 46, 64, 128, tid);               // zero-padded K (46 -> 64)
  load_wt(W2t, w2, 128, 128, 128, tid);
  load_wt(W3t, w3, 128, 128, 64, tid);
  load_vec(b1, g1, 128, tid);
  load_vec(b2, g2, 128, tid);
  load_vec(b3, g3, 64, tid);
  for (int i = tid; i < 64 * 32; i += 256) {
    int n = i >> 5, p = i & 31;
    x_sh[i] = f2bf(x[(size_t)bb * (P_ * N_) + p * N_ + n]);
  }
  for (int i = tid; i < 14 * 32; i += 256) {        // yT[v][s], padded to 32 cols
    int v = i >> 5, j = i & 31;
    y_pad[i] = (j < S_) ? f2bf(y[(size_t)bb * (S_ * NV_) + j * NV_ + v]) : (unsigned short)0;
  }
  for (int i = tid; i < 64 * 64; i += 256) acc_sh[i] = 0.f;
  __syncthreads();

  unsigned short* act1 = stg + wave * (2 * 16 * 136);
  unsigned short* act2 = act1 + 16 * 136;
  int n  = lane & 15;
  int mb = (lane >> 4) << 3;

  for (int t = chunk * 28 + wave; t < chunk * 28 + 28; t += 8) {
    int e  = t * 16 + n;
    int kn = e / 14, vv = e - kn * 14;
    const unsigned short* rowK = x_sh + kn * 32;    // K 0..31
    const unsigned short* rowV = y_pad + vv * 32;   // K 32..63 (padded)
#pragma unroll
    for (int nt = 0; nt < 8; nt++) {
      v8f acc = *(const v8f*)(b1 + nt * 16 + mb);
      acc = wmma_bf16(afrag_w(W1t, 64, nt, 0, lane), bfrag_r(rowK, 0, lane), acc);
      acc = wmma_bf16(afrag_w(W1t, 64, nt, 1, lane), bfrag_r(rowV, 0, lane), acc);
      store_row8_relu(act1 + n * 136 + nt * 16 + mb, acc);
    }
    v8f out[4];
    mlp_l23(act1, act2, W2t, b2, W3t, b3, lane, out);

#pragma unroll
    for (int nt = 0; nt < 4; nt++) {
      float* base = acc_sh + kn * 64 + nt * 16 + mb;
#pragma unroll
      for (int i = 0; i < 8; i++) atomicAdd(base + i, out[nt][i]);
    }
  }
  __syncthreads();
  for (int i = tid; i < 64 * 64; i += 256)
    unsafeAtomicAdd(&epv[(size_t)bb * (N_ * DE_) + i], acc_sh[i]);
}

// =================== kernel C: object MLP + pooled sum ===================
// grid(64): each block owns 128 rows = 2 whole batches; 1 tile per wave; K1=160
__global__ __launch_bounds__(256) void intagger_obj_kernel(
    const float* __restrict__ x,
    const float* __restrict__ epp, const float* __restrict__ epv,
    const float* __restrict__ w1, const float* __restrict__ g1,
    const float* __restrict__ w2, const float* __restrict__ g2,
    const float* __restrict__ w3, const float* __restrict__ g3,
    float* __restrict__ pooled)
{
  extern __shared__ unsigned short smem[];
  unsigned short* W1t    = smem;                    // 128*160
  unsigned short* W2t    = W1t + 128 * 160;         // 128*128
  unsigned short* W3t    = W2t + 128 * 128;         // 64*128
  unsigned short* x_sh   = W3t + 64 * 128;          // [128][32]  (2 batches of xT)
  unsigned short* epe_sh = x_sh + 128 * 32;         // [128][128] (epp | epv per row)
  unsigned short* stg    = epe_sh + 128 * 128;      // 8 * 2*(16*136)
  float* pool_sh = (float*)(stg + 8 * 2 * 16 * 136);// [2][64]
  float* b1 = pool_sh + 128;
  float* b2 = b1 + 128;
  float* b3 = b2 + 128;

  int tid = threadIdx.x, lane = tid & 31, wave = tid >> 5;
  int b0 = blockIdx.x * 2;                          // two whole batches per block

  load_wt(W1t, w1, 160, 160, 128, tid);
  load_wt(W2t, w2, 128, 128, 128, tid);
  load_wt(W3t, w3, 128, 128, 64, tid);
  load_vec(b1, g1, 128, tid);
  load_vec(b2, g2, 128, tid);
  load_vec(b3, g3, 64, tid);
  for (int i = tid; i < 128 * 32; i += 256) {       // xT rows for both batches
    int lr = i >> 5, p = i & 31;
    int lb = lr >> 6, nn = lr & 63;
    x_sh[i] = f2bf(x[((size_t)(b0 + lb)) * (P_ * N_) + p * N_ + nn]);
  }
  for (int i = tid; i < 128 * 128; i += 256) {      // [epp | epv] rows, coalesced
    int lr = i >> 7, c = i & 127;
    size_t rowoff = ((size_t)b0 * N_ + lr) * DE_;
    float v = (c < 64) ? epp[rowoff + c] : epv[rowoff + (c - 64)];
    epe_sh[i] = f2bf(v);
  }
  for (int i = tid; i < 128; i += 256) pool_sh[i] = 0.f;
  __syncthreads();

  unsigned short* act1 = stg + wave * (2 * 16 * 136);
  unsigned short* act2 = act1 + 16 * 136;
  int n  = lane & 15;
  int mb = (lane >> 4) << 3;

  int lrow = wave * 16 + n;                         // 0..127
  const unsigned short* rowX = x_sh + lrow * 32;    // K 0..31
  const unsigned short* rowE = epe_sh + lrow * 128; // K 32..159
#pragma unroll
  for (int nt = 0; nt < 8; nt++) {
    v8f acc = *(const v8f*)(b1 + nt * 16 + mb);
    acc = wmma_bf16(afrag_w(W1t, 160, nt, 0, lane), bfrag_r(rowX, 0, lane), acc);
#pragma unroll
    for (int ks = 1; ks < 5; ks++)
      acc = wmma_bf16(afrag_w(W1t, 160, nt, ks, lane), bfrag_r(rowE, ks - 1, lane), acc);
    store_row8_relu(act1 + n * 136 + nt * 16 + mb, acc);
  }
  v8f out[4];
  mlp_l23(act1, act2, W2t, b2, W3t, b3, lane, out);

  // pooled[b, feat] += row outputs (LDS accumulate, then plain store)
  int lb = lrow >> 6;
#pragma unroll
  for (int nt = 0; nt < 4; nt++) {
    float* base = pool_sh + lb * 64 + nt * 16 + mb;
#pragma unroll
    for (int i = 0; i < 8; i++) atomicAdd(base + i, out[nt][i]);
  }
  __syncthreads();
  for (int i = tid; i < 128; i += 256)
    pooled[(size_t)b0 * DO_ + i] = pool_sh[i];      // block owns these batches
}

// =================== kernel D: final classifier [B,64] @ [64,5] + b ===================
__global__ void intagger_fc_kernel(const float* __restrict__ pooled,
                                   const float* __restrict__ fcw,
                                   const float* __restrict__ fcb,
                                   float* __restrict__ out) {
  int i = blockIdx.x * blockDim.x + threadIdx.x;
  if (i >= B_ * NCLS_) return;
  int b = i / NCLS_, c = i - b * NCLS_;
  float acc = fcb[c];
#pragma unroll
  for (int k = 0; k < DO_; k++) acc += pooled[b * DO_ + k] * fcw[k * NCLS_ + c];
  out[i] = acc;
}

extern "C" void kernel_launch(void* const* d_in, const int* in_sizes, int n_in,
                              void* d_out, int out_size, void* d_ws, size_t ws_size,
                              hipStream_t stream) {
  const float* x       = (const float*)d_in[0];
  const float* y       = (const float*)d_in[1];
  const float* fr1_w   = (const float*)d_in[2];
  const float* fr1_b   = (const float*)d_in[3];
  const float* fr2_w   = (const float*)d_in[4];
  const float* fr2_b   = (const float*)d_in[5];
  const float* fr3_w   = (const float*)d_in[6];
  const float* fr3_b   = (const float*)d_in[7];
  const float* fr1pv_w = (const float*)d_in[8];
  const float* fr1pv_b = (const float*)d_in[9];
  const float* fr2pv_w = (const float*)d_in[10];
  const float* fr2pv_b = (const float*)d_in[11];
  const float* fr3pv_w = (const float*)d_in[12];
  const float* fr3pv_b = (const float*)d_in[13];
  const float* fo1_w   = (const float*)d_in[14];
  const float* fo1_b   = (const float*)d_in[15];
  const float* fo2_w   = (const float*)d_in[16];
  const float* fo2_b   = (const float*)d_in[17];
  const float* fo3_w   = (const float*)d_in[18];
  const float* fo3_b   = (const float*)d_in[19];
  const float* fc_w    = (const float*)d_in[20];
  const float* fc_b    = (const float*)d_in[21];

  // workspace: Ebar_pp [B,N,DE], Ebar_pv [B,N,DE], pooled [B,DO] (~4.03 MB)
  float* epp    = (float*)d_ws;
  float* epv    = epp + (size_t)B_ * N_ * DE_;
  float* pooled = epv + (size_t)B_ * N_ * DE_;

  int zn = B_ * N_ * DE_ * 2;                       // pooled fully overwritten by obj kernel
  intagger_zero_kernel<<<(zn + 255) / 256, 256, 0, stream>>>(epp, zn);

  // dynamic LDS (bytes)
  size_t smem_pp  = (size_t)(128*64 + 128*128 + 64*128 + 64*32 + 8*2*16*136) * 2
                  + (size_t)(64*64 + 320) * 4;
  size_t smem_pv  = (size_t)(128*64 + 128*128 + 64*128 + 64*32 + 512 + 8*2*16*136) * 2
                  + (size_t)(64*64 + 320) * 4;
  size_t smem_obj = (size_t)(128*160 + 128*128 + 64*128 + 128*32 + 128*128 + 8*2*16*136) * 2
                  + (size_t)(128 + 320) * 4;

  intagger_pp_kernel<<<dim3(B_, 4), 256, smem_pp, stream>>>(
      x, fr1_w, fr1_b, fr2_w, fr2_b, fr3_w, fr3_b, epp);

  intagger_pv_kernel<<<dim3(B_, 2), 256, smem_pv, stream>>>(
      x, y, fr1pv_w, fr1pv_b, fr2pv_w, fr2pv_b, fr3pv_w, fr3pv_b, epv);

  intagger_obj_kernel<<<dim3(64), 256, smem_obj, stream>>>(
      x, epp, epv, fo1_w, fo1_b, fo2_w, fo2_b, fo3_w, fo3_b, pooled);

  intagger_fc_kernel<<<(B_ * NCLS_ + 255) / 256, 256, 0, stream>>>(
      pooled, fc_w, fc_b, (float*)d_out);
}
